// MyRGCNConvHistory_39522289058165
// MI455X (gfx1250) — matrix operand order repaired and checked by reading, not compiled, not run on hardware
//
#include <hip/hip_runtime.h>

typedef __attribute__((ext_vector_type(16))) __bf16       v16bf;
typedef __attribute__((ext_vector_type(8)))  float        v8f;
typedef __attribute__((ext_vector_type(4)))  unsigned int u32x4;
typedef __attribute__((ext_vector_type(2)))  unsigned int u32x2;
typedef __attribute__((ext_vector_type(4)))  float        f32x4;

#define IN_C   128
#define HID_C  128
#define NREL   8
#define NPB    32          // nodes per block (two 16-row M-tiles)

static __device__ __forceinline__ unsigned int pack2_bf16(float a, float b) {
    union { float f; unsigned int u; } x, y;
    x.f = a; y.f = b;
    unsigned int ra = x.u + 0x7FFFu + ((x.u >> 16) & 1u);   // round-to-nearest-even
    unsigned int rb = y.u + 0x7FFFu + ((y.u >> 16) & 1u);
    return (ra >> 16) | (rb & 0xFFFF0000u);
}

// rel_weight f32 [r][k_in][n_hid]  ->  Wb bf16 [r][n_hid][k_in]  (transposed, K contiguous)
__global__ void rgcn_convert_weights(const float* __restrict__ W,
                                     unsigned short* __restrict__ Wb) {
    int t = blockIdx.x * blockDim.x + threadIdx.x;        // one thread per (r,n,k-pair)
    if (t >= NREL * HID_C * (IN_C / 2)) return;
    int k2 = (t & (IN_C / 2 - 1)) * 2;
    int n  = (t / (IN_C / 2)) & (HID_C - 1);
    int r  = t / ((IN_C / 2) * HID_C);
    float a = W[((size_t)r * IN_C + k2    ) * HID_C + n];
    float b = W[((size_t)r * IN_C + k2 + 1) * HID_C + n];
    *(unsigned int*)&Wb[((size_t)r * HID_C + n) * IN_C + k2] = pack2_bf16(a, b);
}

__global__ __launch_bounds__(256)
void rgcn_history_kernel(const float* __restrict__ x,
                         const unsigned short* __restrict__ Wb,
                         const int* __restrict__ ptr,
                         const int* __restrict__ idx,
                         const int* __restrict__ et,
                         const int* __restrict__ hmap,
                         const float* __restrict__ hbuf,
                         const int* __restrict__ hsize_p,
                         float* __restrict__ out,
                         int num_node) {
    __shared__ unsigned short sA[NREL * NPB * IN_C];  // 64 KB: bf16 per-relation aggregates
    __shared__ float sRDeg[NPB];
    __shared__ int   sHist[NPB];

    const int tid  = threadIdx.x;
    const int lane = tid & 31;
    const int wave = tid >> 5;
    const int base = blockIdx.x * NPB;

    // ---------------- Phase 1: per-(node, relation) neighbor aggregation ----------------
    // wave w handles local nodes 4w..4w+3; each lane owns 4 of 128 channels.
    for (int nn = 0; nn < NPB / 8; ++nn) {
        const int nl = wave * (NPB / 8) + nn;
        const int g  = base + nl;
        f32x4 acc[NREL];
#pragma unroll
        for (int r = 0; r < NREL; ++r) acc[r] = (f32x4){0.f, 0.f, 0.f, 0.f};

        if (g < num_node) {
            const int e0 = ptr[g], e1 = ptr[g + 1];
            if (lane == 0) {
                sRDeg[nl] = (e1 > e0) ? 1.0f / (float)(e1 - e0) : 0.0f;
                sHist[nl] = hmap[g];
            }
            const int col = lane * 4;
            for (int e = e0; e < e1; ++e) {
                // edge data is wave-uniform -> scalarize
                const int src = __builtin_amdgcn_readfirstlane(idx[e]);
                const int r   = __builtin_amdgcn_readfirstlane(et[e]);
                const f32x4 v = *(const f32x4*)(x + (size_t)src * IN_C + col);
                switch (r) {                      // scalar branch: 4 v_add_f32 per edge
                    case 0: acc[0] += v; break;
                    case 1: acc[1] += v; break;
                    case 2: acc[2] += v; break;
                    case 3: acc[3] += v; break;
                    case 4: acc[4] += v; break;
                    case 5: acc[5] += v; break;
                    case 6: acc[6] += v; break;
                    default: acc[7] += v; break;
                }
            }
        } else if (lane == 0) {
            sRDeg[nl] = 0.0f;
            sHist[nl] = -1;
        }

        // store bf16 aggregates, row-major [r][m][k] (A fragment = 2x ds_load_b128)
        unsigned short* p = &sA[nl * IN_C + lane * 4];
#pragma unroll
        for (int r = 0; r < NREL; ++r) {
            u32x2 t;
            t.x = pack2_bf16(acc[r].x, acc[r].y);
            t.y = pack2_bf16(acc[r].z, acc[r].w);
            *(u32x2*)(p + r * NPB * IN_C) = t;
        }
    }
    __syncthreads();

    // ---------------- Phase 2: WMMA  D[32 x 128] = sum_r A_r[32x128] @ W_r[128x128] ----
    // wave w owns output columns [16w, 16w+16). Each B fragment is loaded once and
    // reused for both 16-row M-tiles: 8 rel * 4 K-chunks * 2 tiles = 64 WMMAs/wave.
    const int m0     = lane & 15;
    const int khalfA = (lane < 16) ? 0 : 16;   // A k order used here (same as B): lanes 0-15 K0-15, hi K16-31
    const int khalfB = (lane < 16) ? 0 : 16;   // B k order used here (same as A): lanes 0-15 K0-15, hi K16-31
    const int ncol   = wave * 16 + m0;

    v8f c0 = {0.f, 0.f, 0.f, 0.f, 0.f, 0.f, 0.f, 0.f};
    v8f c1 = {0.f, 0.f, 0.f, 0.f, 0.f, 0.f, 0.f, 0.f};
    for (int r = 0; r < NREL; ++r) {
        const unsigned short* ap0 = &sA[(r * NPB + m0) * IN_C];        // M-tile 0
        const unsigned short* ap1 = &sA[(r * NPB + 16 + m0) * IN_C];   // M-tile 1
        const unsigned short* bp  = Wb + ((size_t)r * HID_C + ncol) * IN_C;
#pragma unroll
        for (int kc = 0; kc < 4; ++kc) {
            union { v16bf v; u32x4 q[2]; } a0, a1, bf;
            const int kb = kc * 32 + khalfB;
            bf.q[0] = *(const u32x4*)(bp + kb);        // global_load_b128 (loaded once)
            bf.q[1] = *(const u32x4*)(bp + kb + 8);
            const int ka = kc * 32 + khalfA;
            a0.q[0] = *(const u32x4*)(ap0 + ka);       // ds_load_b128
            a0.q[1] = *(const u32x4*)(ap0 + ka + 8);
            a1.q[0] = *(const u32x4*)(ap1 + ka);
            a1.q[1] = *(const u32x4*)(ap1 + ka + 8);
            c0 = __builtin_amdgcn_wmma_f32_16x16x32_bf16(
                     false, a0.v, false, bf.v, (short)0, c0, false, false);
            c1 = __builtin_amdgcn_wmma_f32_16x16x32_bf16(
                     false, a1.v, false, bf.v, (short)0, c1, false, false);
        }
    }

    // ---------------- Phase 3: normalize, history override, store both tuple halves ----
    const int    hsize = hsize_p[0];
    const size_t tot   = (size_t)num_node * HID_C;
#pragma unroll
    for (int i = 0; i < 8; ++i) {
        const int mr = (lane < 16) ? i : (i + 8);      // C/D layout: VGPR i -> row
        {   // M-tile 0
            const int g = base + mr;
            if (g < num_node) {
                float val = c0[i] * sRDeg[mr];
                if (hsize > 0 && sHist[mr] != -1)
                    val = hbuf[(size_t)g * HID_C + ncol];
                out[(size_t)g * HID_C + ncol]       = val;
                out[tot + (size_t)g * HID_C + ncol] = val;
            }
        }
        {   // M-tile 1
            const int m = 16 + mr;
            const int g = base + m;
            if (g < num_node) {
                float val = c1[i] * sRDeg[m];
                if (hsize > 0 && sHist[m] != -1)
                    val = hbuf[(size_t)g * HID_C + ncol];
                out[(size_t)g * HID_C + ncol]       = val;
                out[tot + (size_t)g * HID_C + ncol] = val;
            }
        }
    }
}

extern "C" void kernel_launch(void* const* d_in, const int* in_sizes, int n_in,
                              void* d_out, int out_size, void* d_ws, size_t ws_size,
                              hipStream_t stream) {
    const float* x    = (const float*)d_in[0];
    const float* W    = (const float*)d_in[1];
    const int*   ptr  = (const int*)d_in[2];
    const int*   idx  = (const int*)d_in[3];
    const int*   et   = (const int*)d_in[4];
    const int*   hmap = (const int*)d_in[5];
    const float* hbuf = (const float*)d_in[6];
    const int*   hsz  = (const int*)d_in[7];

    const int num_node = in_sizes[0] / IN_C;

    unsigned short* Wb = (unsigned short*)d_ws;        // 8*128*128*2 = 256 KB

    const int cvt_threads = NREL * HID_C * (IN_C / 2); // 65536
    rgcn_convert_weights<<<(cvt_threads + 255) / 256, 256, 0, stream>>>(W, Wb);

    const int nblk = (num_node + NPB - 1) / NPB;       // 3125 blocks of 32 nodes
    rgcn_history_kernel<<<nblk, 256, 0, stream>>>(x, Wb, ptr, idx, et, hmap, hbuf,
                                                  hsz, (float*)d_out, num_node);
}
